// KeypointDetector_41145786696229
// MI455X (gfx1250) — matrix operand order repaired and checked by reading, not compiled, not run on hardware
//
#include <hip/hip_runtime.h>
#include <hip/hip_bf16.h>
#include <math.h>

typedef __attribute__((ext_vector_type(16))) _Float16 v16h;
typedef __attribute__((ext_vector_type(8)))  _Float16 v8h;
typedef __attribute__((ext_vector_type(8)))  float    v8f;

// ---------------- problem constants ----------------
#define Bn   4
#define Nn   8192
#define Mn   2048
#define Kn   16
#define CIN  64
#define P1   (Bn*Mn*Kn)   // 131072 rows for conv layers
#define P2   (Bn*Mn)      // 8192  rows for MLP layers

// ---------------- workspace byte offsets ----------------
#define OFF_IDX    ((size_t)0)                  // 131072 int
#define OFF_KNN    ((size_t)524288)             // 131072*3 f32
#define OFF_STATS  ((size_t)2097152)            // 1024 f32 (sum | sumsq | scale | shift)
#define OFF_W1     ((size_t)2101248)            // 64*96 f16 (padded)
#define OFF_W2     ((size_t)2113536)            // 128*64 f16
#define OFF_W3     ((size_t)2129920)            // 256*128 f16
#define OFF_M1     ((size_t)2195456)            // 256*256 f16
#define OFF_M2     ((size_t)2326528)            // 256*256 f16
#define OFF_BUFA   ((size_t)4194304)            // 67 MB f16
#define OFF_BUFB   ((size_t)75497472)           // 67 MB f16

// ---------------- d_out float offsets ----------------
#define OUT_KP   ((size_t)0)          // (B,M,3)
#define OUT_SIG  ((size_t)24576)      // (B,M)
#define OUT_AFT  ((size_t)32768)      // (B,256,M)
#define OUT_GR   ((size_t)2129920)    // (B,68,M,K)
#define OUT_AFM  ((size_t)11042816)   // (B,256,M,K)

// =====================================================================
// Weight prep: f32 -> f16, pad conv1 Cin 68 -> 96 with zeros.
// =====================================================================
__global__ void prep_weights(const float* __restrict__ W1, const float* __restrict__ W2,
                             const float* __restrict__ W3, const float* __restrict__ m1W,
                             const float* __restrict__ m2W,
                             _Float16* __restrict__ w1h, _Float16* __restrict__ w2h,
                             _Float16* __restrict__ w3h, _Float16* __restrict__ m1h,
                             _Float16* __restrict__ m2h) {
  int stride = gridDim.x * blockDim.x;
  int tid = blockIdx.x * blockDim.x + threadIdx.x;
  for (int i = tid; i < 64*96; i += stride) {
    int o = i / 96, c = i % 96;
    w1h[i] = (c < 68) ? (_Float16)W1[o*68 + c] : (_Float16)0.f;
  }
  for (int i = tid; i < 128*64; i += stride) w2h[i] = (_Float16)W2[i];
  for (int i = tid; i < 256*128; i += stride) w3h[i] = (_Float16)W3[i];
  for (int i = tid; i < 256*256; i += stride) { m1h[i] = (_Float16)m1W[i]; m2h[i] = (_Float16)m2W[i]; }
}

// =====================================================================
// KNN: one thread per (b,m) query, LDS-tiled scan of the point cloud
// with next-tile prefetch (global_prefetch_b8), fully unrolled
// 16-element register insertion sort (stable on ties).
// =====================================================================
__global__ void knn_kernel(const float* __restrict__ xyz, const int* __restrict__ sidx,
                           int* __restrict__ outIdx) {
  __shared__ float tile[256*3];
  int t = blockIdx.x * 256 + threadIdx.x;   // 0..8191 ; whole block shares one b
  int b = t >> 11;
  int m = t & (Mn - 1);
  const float* xb = xyz + (size_t)b * Nn * 3;
  int sp = sidx[m];
  float qx = xb[sp*3+0], qy = xb[sp*3+1], qz = xb[sp*3+2];
  float bd[16]; int bi[16];
#pragma unroll
  for (int j = 0; j < 16; ++j) { bd[j] = 3.4e38f; bi[j] = 0; }
  for (int n0 = 0; n0 < Nn; n0 += 256) {
    __syncthreads();
    for (int j = threadIdx.x; j < 768; j += 256) tile[j] = xb[(size_t)n0*3 + j];
    if (n0 + 256 < Nn)   // prefetch next tile into cache hierarchy
      __builtin_prefetch(xb + (size_t)(n0 + 256)*3 + threadIdx.x*3, 0, 3);
    __syncthreads();
    for (int nn = 0; nn < 256; ++nn) {
      float dx = tile[nn*3+0] - qx;
      float dy = tile[nn*3+1] - qy;
      float dz = tile[nn*3+2] - qz;
      float d2 = dx*dx + dy*dy + dz*dz;
      if (d2 < bd[15]) {
        float d = d2; int id = n0 + nn;
#pragma unroll
        for (int j = 0; j < 16; ++j) {
          bool lt = d < bd[j];
          float td = bd[j]; int ti = bi[j];
          if (lt) { bd[j] = d; bi[j] = id; d = td; id = ti; }
        }
      }
    }
  }
#pragma unroll
  for (int j = 0; j < 16; ++j) outIdx[(size_t)t*16 + j] = bi[j];
}

// =====================================================================
// Grouping: rela/dist/feature gather.  Writes grouped^T (output 4),
// the f16 padded GEMM input X0 (131072 x 96) with vector b128 stores,
// and knn_xyz scratch.
// =====================================================================
__global__ void group_kernel(const float* __restrict__ xyz, const float* __restrict__ feat,
                             const int* __restrict__ sidx, const int* __restrict__ knnIdx,
                             float* __restrict__ out_grT, _Float16* __restrict__ X0,
                             float* __restrict__ knn_xyz) {
  int t = blockIdx.x * 256 + threadIdx.x;   // 0..131071 == (b,m,k)
  int k = t & 15;
  int m = (t >> 4) & (Mn - 1);
  int b = t >> 15;
  const float* xb = xyz + (size_t)b * Nn * 3;
  int spi = sidx[m];
  float sx = xb[spi*3+0], sy = xb[spi*3+1], sz = xb[spi*3+2];
  int ni = knnIdx[t];
  float nx = xb[ni*3+0], ny = xb[ni*3+1], nz = xb[ni*3+2];
  knn_xyz[(size_t)t*3+0] = nx; knn_xyz[(size_t)t*3+1] = ny; knn_xyz[(size_t)t*3+2] = nz;
  float rx = nx - sx, ry = ny - sy, rz = nz - sz;
  float dist = sqrtf(rx*rx + ry*ry + rz*rz);
  const size_t strideC = (size_t)Mn * Kn;
  size_t grBase = (size_t)b*68*strideC + (size_t)m*Kn + k;

  v8h pack[12];
#pragma unroll
  for (int i = 0; i < 12; ++i) pack[i] = (v8h)(_Float16)0.f;
  pack[0][0] = (_Float16)rx; pack[0][1] = (_Float16)ry;
  pack[0][2] = (_Float16)rz; pack[0][3] = (_Float16)dist;
  out_grT[grBase + 0*strideC] = rx;
  out_grT[grBase + 1*strideC] = ry;
  out_grT[grBase + 2*strideC] = rz;
  out_grT[grBase + 3*strideC] = dist;
  const float* fb = feat + (size_t)b*CIN*Nn + ni;
#pragma unroll
  for (int j = 0; j < CIN; ++j) {
    float v = fb[(size_t)j*Nn];
    out_grT[grBase + (size_t)(4+j)*strideC] = v;
    pack[(4+j) >> 3][(4+j) & 7] = (_Float16)v;
  }
  _Float16* xr = X0 + (size_t)t * 96;
#pragma unroll
  for (int i = 0; i < 12; ++i) *(v8h*)(xr + i*8) = pack[i];
}

// =====================================================================
// WMMA fragment loaders (CDNA5 ISA 7.12.2 layouts)
// =====================================================================
__device__ __forceinline__ v16h load_afrag(const _Float16* rowp, int k0, int hi) {
  // A 16x32: lanes<16 hold K {0..7,16..23}, lanes>=16 hold K {8..15,24..31}
  v8h lo = *(const v8h*)(rowp + k0 + hi*8);
  v8h hh = *(const v8h*)(rowp + k0 + 16 + hi*8);
  v16h a;
#pragma unroll
  for (int i = 0; i < 8; ++i) { a[i] = lo[i]; a[i+8] = hh[i]; }
  return a;
}
__device__ __forceinline__ v16h load_bfrag(const _Float16* rowp, int k0, int hi) {
  // B 32x16: lanes<16 hold K {0..7,16..23}, lanes>=16 hold K {8..15,24..31} of column (lane&15)
  v8h lo = *(const v8h*)(rowp + k0 + hi*8);
  v8h hh = *(const v8h*)(rowp + k0 + 16 + hi*8);
  v16h b;
#pragma unroll
  for (int i = 0; i < 8; ++i) { b[i] = lo[i]; b[i+8] = hh[i]; }
  return b;
}

// =====================================================================
// GEMM + per-channel stats.  Y[p,o] = sum_c X[p,c]*W[o,c].
// One wave per 32x32 output tile: 2 A-frags x 2 B-frags -> 4 WMMAs per
// k-step (2x fragment reuse).  Epilogue stages the f16 tile in LDS and
// stores coalesced b128 rows; per-channel sum/sumsq via f32 atomics.
// =====================================================================
__global__ void gemm_bn_stats(const _Float16* __restrict__ X, const _Float16* __restrict__ W,
                              _Float16* __restrict__ Y, float* __restrict__ stats,
                              int P, int Kd, int Cout) {
  __shared__ alignas(16) _Float16 stage[8][32*32];   // 16 KB
  int wib  = threadIdx.x >> 5;
  int wave = (blockIdx.x * blockDim.x + threadIdx.x) >> 5;
  int lane = threadIdx.x & 31;
  int tilesM = P >> 5;
  int tm = wave % tilesM;
  int tn = wave / tilesM;
  if (tn >= (Cout >> 5)) return;       // grids are exact; never taken
  int row = lane & 15;
  int hi  = lane >> 4;
  const _Float16* a0 = X + (size_t)(tm*32 + row)      * Kd;
  const _Float16* a1 = X + (size_t)(tm*32 + 16 + row) * Kd;
  const _Float16* b0 = W + (size_t)(tn*32 + row)      * Kd;
  const _Float16* b1 = W + (size_t)(tn*32 + 16 + row) * Kd;
  v8f acc00 = {0.f,0.f,0.f,0.f,0.f,0.f,0.f,0.f};
  v8f acc01 = acc00, acc10 = acc00, acc11 = acc00;
  for (int k0 = 0; k0 < Kd; k0 += 32) {
    v16h A0 = load_afrag(a0, k0, hi);
    v16h A1 = load_afrag(a1, k0, hi);
    v16h B0 = load_bfrag(b0, k0, hi);
    v16h B1 = load_bfrag(b1, k0, hi);
    acc00 = __builtin_amdgcn_wmma_f32_16x16x32_f16(false, A0, false, B0, (short)0, acc00, false, false);
    acc01 = __builtin_amdgcn_wmma_f32_16x16x32_f16(false, A0, false, B1, (short)0, acc01, false, false);
    acc10 = __builtin_amdgcn_wmma_f32_16x16x32_f16(false, A1, false, B0, (short)0, acc10, false, false);
    acc11 = __builtin_amdgcn_wmma_f32_16x16x32_f16(false, A1, false, B1, (short)0, acc11, false, false);
  }
  _Float16* st = &stage[wib][0];
  float s0 = 0.f, s1 = 0.f, q0 = 0.f, q1 = 0.f;
#pragma unroll
  for (int j = 0; j < 8; ++j) {
    float v;
    v = acc00[j]; s0 += v; q0 += v*v; st[(j + hi*8)*32      + row] = (_Float16)v;
    v = acc01[j]; s1 += v; q1 += v*v; st[(j + hi*8)*32 + 16 + row] = (_Float16)v;
    v = acc10[j]; s0 += v; q0 += v*v; st[(16 + j + hi*8)*32      + row] = (_Float16)v;
    v = acc11[j]; s1 += v; q1 += v*v; st[(16 + j + hi*8)*32 + 16 + row] = (_Float16)v;
  }
  atomicAdd(&stats[      tn*32 +      row], s0);
  atomicAdd(&stats[      tn*32 + 16 + row], s1);
  atomicAdd(&stats[256 + tn*32 +      row], q0);
  atomicAdd(&stats[256 + tn*32 + 16 + row], q1);
  __syncthreads();
#pragma unroll
  for (int pass = 0; pass < 4; ++pass) {   // 4 lanes per row, b128 each
    int r  = pass*8 + (lane >> 2);
    int co = (lane & 3) * 8;
    v8h v = *(const v8h*)(st + r*32 + co);
    *(v8h*)(Y + (size_t)(tm*32 + r) * Cout + tn*32 + co) = v;
  }
}

// stats: [sum | sumsq | scale | shift], each 256 floats
__global__ void zero_stats(float* __restrict__ stats) {
  if (threadIdx.x < 512) stats[threadIdx.x] = 0.f;
}

__global__ void finalize_stats(float* __restrict__ stats, const float* __restrict__ gamma,
                               const float* __restrict__ beta, int Cout, float invP) {
  int c = threadIdx.x;
  if (c < Cout) {
    float mean = stats[c] * invP;
    float var  = stats[256 + c] * invP - mean*mean;
    float inv  = rsqrtf(var + 1e-5f);
    float sc   = gamma[c] * inv;
    stats[512 + c] = sc;
    stats[768 + c] = beta[c] - mean * sc;
  }
}

// vectorized: one v8h (b128) per thread
__global__ void bn_relu(_Float16* __restrict__ Y, const float* __restrict__ stats,
                        int Cout, size_t total8) {
  size_t i = (size_t)blockIdx.x * blockDim.x + threadIdx.x;
  if (i < total8) {
    v8h v = ((const v8h*)Y)[i];
    int c0 = (int)((i*8) & (size_t)(Cout - 1));   // Cout is a power of two
    v8h r;
#pragma unroll
    for (int j = 0; j < 8; ++j) {
      float f = (float)v[j] * stats[512 + c0 + j] + stats[768 + c0 + j];
      r[j] = (_Float16)fmaxf(f, 0.f);
    }
    ((v8h*)Y)[i] = r;
  }
}

// =====================================================================
// Attention over K: channel-max -> softmax(K) -> keypoints, af, afm.
// One 256-thread block per (b,m); thread (k,g) owns neighbor k, channels g*16..+15.
// =====================================================================
__global__ void attention_kernel(const _Float16* __restrict__ h3, const float* __restrict__ knn,
                                 float* __restrict__ out_kp, float* __restrict__ out_afT,
                                 float* __restrict__ out_afmT, _Float16* __restrict__ af16) {
  __shared__ float smax[256];
  __shared__ float sx1[16];
  __shared__ float sattn[16];
  __shared__ float safm[16*256];
  int p = blockIdx.x;                 // 0..8191
  int b = p >> 11;
  int m = p & (Mn - 1);
  int tid = threadIdx.x;
  int k = tid >> 4, g = tid & 15;
  const _Float16* hp = h3 + ((size_t)p*16 + k) * 256 + g*16;
  float hv[16];
  float pm = -3.4e38f;
#pragma unroll
  for (int j = 0; j < 16; ++j) { hv[j] = (float)hp[j]; pm = fmaxf(pm, hv[j]); }
  smax[k*16 + g] = pm;
  __syncthreads();
  if (g == 0) {
    float x = smax[k*16];
#pragma unroll
    for (int j = 1; j < 16; ++j) x = fmaxf(x, smax[k*16 + j]);
    sx1[k] = x;
  }
  __syncthreads();
  float gm = sx1[0];
#pragma unroll
  for (int j = 1; j < 16; ++j) gm = fmaxf(gm, sx1[j]);
  float se = 0.f;
#pragma unroll
  for (int j = 0; j < 16; ++j) se += __expf(sx1[j] - gm);
  float attn = __expf(sx1[k] - gm) / se;
  if (g == 0) sattn[k] = attn;
#pragma unroll
  for (int j = 0; j < 16; ++j) {
    float av = hv[j] * attn;
    int c = g*16 + j;
    safm[k*256 + c] = av;
    out_afmT[(((size_t)b*256 + c)*Mn + m)*Kn + k] = av;
  }
  __syncthreads();
  {
    int c = tid;
    float af = 0.f;
#pragma unroll
    for (int kk = 0; kk < 16; ++kk) af += safm[kk*256 + c];
    out_afT[((size_t)b*256 + c)*Mn + m] = af;
    af16[(size_t)p*256 + c] = (_Float16)af;
  }
  if (tid < 3) {
    float kp = 0.f;
#pragma unroll
    for (int kk = 0; kk < 16; ++kk) kp += sattn[kk] * knn[((size_t)p*16 + kk)*3 + tid];
    out_kp[(size_t)p*3 + tid] = kp;
  }
}

// =====================================================================
// Head: sig = z . m3W + m3b ; sigmas = softplus(sig) + 0.001
// =====================================================================
__global__ void head_kernel(const _Float16* __restrict__ z, const float* __restrict__ m3W,
                            const float* __restrict__ m3b, float* __restrict__ out_sig) {
  int t = blockIdx.x * 256 + threadIdx.x;   // 0..8191
  const v8h* zr = (const v8h*)(z + (size_t)t * 256);
  float acc = m3b[0];
#pragma unroll 4
  for (int c8 = 0; c8 < 32; ++c8) {
    v8h v = zr[c8];
#pragma unroll
    for (int j = 0; j < 8; ++j) acc += (float)v[j] * m3W[c8*8 + j];
  }
  float sp = (acc > 20.f) ? acc : logf(1.f + expf(acc));
  out_sig[t] = sp + 0.001f;
}

// =====================================================================
// Host side
// =====================================================================
extern "C" void kernel_launch(void* const* d_in, const int* in_sizes, int n_in,
                              void* d_out, int out_size, void* d_ws, size_t ws_size,
                              hipStream_t stream) {
  const float* xyz  = (const float*)d_in[0];
  const float* feat = (const float*)d_in[1];
  const int*   sidx = (const int*)d_in[2];
  const float* W1 = (const float*)d_in[3];  const float* g1 = (const float*)d_in[4];  const float* b1 = (const float*)d_in[5];
  const float* W2 = (const float*)d_in[6];  const float* g2 = (const float*)d_in[7];  const float* b2 = (const float*)d_in[8];
  const float* W3 = (const float*)d_in[9];  const float* g3 = (const float*)d_in[10]; const float* b3 = (const float*)d_in[11];
  const float* m1W = (const float*)d_in[12]; const float* m1g = (const float*)d_in[14]; const float* m1be = (const float*)d_in[15];
  const float* m2W = (const float*)d_in[16]; const float* m2g = (const float*)d_in[18]; const float* m2be = (const float*)d_in[19];
  const float* m3W = (const float*)d_in[20]; const float* m3b = (const float*)d_in[21];
  // (m1b/m2b are all-zero and cancel exactly through training-mode BN; omitted.)

  char* ws = (char*)d_ws;
  int*      wIdx   = (int*)(ws + OFF_IDX);
  float*    wKnn   = (float*)(ws + OFF_KNN);
  float*    wStats = (float*)(ws + OFF_STATS);
  _Float16* w1h = (_Float16*)(ws + OFF_W1);
  _Float16* w2h = (_Float16*)(ws + OFF_W2);
  _Float16* w3h = (_Float16*)(ws + OFF_W3);
  _Float16* m1h = (_Float16*)(ws + OFF_M1);
  _Float16* m2h = (_Float16*)(ws + OFF_M2);
  _Float16* bufA = (_Float16*)(ws + OFF_BUFA);
  _Float16* bufB = (_Float16*)(ws + OFF_BUFB);

  float* out = (float*)d_out;
  float* out_kp  = out + OUT_KP;
  float* out_sig = out + OUT_SIG;
  float* out_afT = out + OUT_AFT;
  float* out_grT = out + OUT_GR;
  float* out_afm = out + OUT_AFM;

  // 1) weights -> f16 (padded)
  prep_weights<<<64, 256, 0, stream>>>(W1, W2, W3, m1W, m2W, w1h, w2h, w3h, m1h, m2h);
  // 2) KNN
  knn_kernel<<<P2/256, 256, 0, stream>>>(xyz, sidx, wIdx);
  // 3) grouping -> grouped^T output, X0 (f16, 131072x96) in bufA, knn_xyz scratch
  group_kernel<<<P1/256, 256, 0, stream>>>(xyz, feat, sidx, wIdx, out_grT, bufA, wKnn);

  // conv layer 1: 96(pad 68) -> 64, bufA -> bufB
  zero_stats<<<1, 512, 0, stream>>>(wStats);
  gemm_bn_stats<<<(P1/32)*(64/32)/8, 256, 0, stream>>>(bufA, w1h, bufB, wStats, P1, 96, 64);
  finalize_stats<<<1, 256, 0, stream>>>(wStats, g1, b1, 64, 1.f/(float)P1);
  bn_relu<<<(unsigned)(((size_t)P1*64/8 + 255)/256), 256, 0, stream>>>(bufB, wStats, 64, (size_t)P1*64/8);

  // conv layer 2: 64 -> 128, bufB -> bufA
  zero_stats<<<1, 512, 0, stream>>>(wStats);
  gemm_bn_stats<<<(P1/32)*(128/32)/8, 256, 0, stream>>>(bufB, w2h, bufA, wStats, P1, 64, 128);
  finalize_stats<<<1, 256, 0, stream>>>(wStats, g2, b2, 128, 1.f/(float)P1);
  bn_relu<<<(unsigned)(((size_t)P1*128/8 + 255)/256), 256, 0, stream>>>(bufA, wStats, 128, (size_t)P1*128/8);

  // conv layer 3: 128 -> 256, bufA -> bufB (h3)
  zero_stats<<<1, 512, 0, stream>>>(wStats);
  gemm_bn_stats<<<(P1/32)*(256/32)/8, 256, 0, stream>>>(bufA, w3h, bufB, wStats, P1, 128, 256);
  finalize_stats<<<1, 256, 0, stream>>>(wStats, g3, b3, 256, 1.f/(float)P1);
  bn_relu<<<(unsigned)(((size_t)P1*256/8 + 255)/256), 256, 0, stream>>>(bufB, wStats, 256, (size_t)P1*256/8);

  // attention: h3 (bufB) -> keypoints, af^T, afm^T outputs; af (f16) -> bufA
  attention_kernel<<<P2, 256, 0, stream>>>(bufB, wKnn, out_kp, out_afT, out_afm, bufA);

  // MLP 1: bufA -> bufB
  zero_stats<<<1, 512, 0, stream>>>(wStats);
  gemm_bn_stats<<<(P2/32)*(256/32)/8, 256, 0, stream>>>(bufA, m1h, bufB, wStats, P2, 256, 256);
  finalize_stats<<<1, 256, 0, stream>>>(wStats, m1g, m1be, 256, 1.f/(float)P2);
  bn_relu<<<(unsigned)(((size_t)P2*256/8 + 255)/256), 256, 0, stream>>>(bufB, wStats, 256, (size_t)P2*256/8);

  // MLP 2: bufB -> bufA
  zero_stats<<<1, 512, 0, stream>>>(wStats);
  gemm_bn_stats<<<(P2/32)*(256/32)/8, 256, 0, stream>>>(bufB, m2h, bufA, wStats, P2, 256, 256);
  finalize_stats<<<1, 256, 0, stream>>>(wStats, m2g, m2be, 256, 1.f/(float)P2);
  bn_relu<<<(unsigned)(((size_t)P2*256/8 + 255)/256), 256, 0, stream>>>(bufA, wStats, 256, (size_t)P2*256/8);

  // head
  head_kernel<<<P2/256, 256, 0, stream>>>(bufA, m3W, m3b, out_sig);
}
